// TIRGWithSCAN_2662879723582
// MI455X (gfx1250) — matrix operand order repaired and checked by reading, not compiled, not run on hardware
//
#include <hip/hip_runtime.h>
#include <hip/hip_bf16.h>
#include <math.h>

// Problem dims (fixed by the reference)
#define B_N 256
#define Q_N 256
#define S_N 512
#define D_N 1024
#define EPS_F 1e-8f
#define SLOPE 0.1f
#define SMOOTH 9.0f

typedef __bf16 bf16_t;
typedef bf16_t v2bf  __attribute__((ext_vector_type(2)));
typedef bf16_t v4bf  __attribute__((ext_vector_type(4)));
typedef bf16_t v8bf  __attribute__((ext_vector_type(8)));
typedef bf16_t v16bf __attribute__((ext_vector_type(16)));
typedef float  v8f   __attribute__((ext_vector_type(8)));

// LDS row stride in bf16 units: 32 data + 8 pad = 80 bytes (16B aligned rows,
// odd-ish bank stride to cut ds_load_b128 conflicts).
#define LDS_STRIDE 40

// ---------------------------------------------------------------------------
// WMMA tile engine shared by both GEMMs.
// A-fragment (16x32 bf16): lane half lh picks K {lh*8..+7, 16+lh*8..+7}.
// B-fragment (32x16 bf16): lane half lh picks K {lh*8..+7, 16+lh*8..+7}.
// ---------------------------------------------------------------------------
__device__ __forceinline__ v16bf make_frag(const bf16_t* row, int off0, int off1) {
  v8bf lo = *(const v8bf*)(row + off0);
  v8bf hi = *(const v8bf*)(row + off1);
  return __builtin_shufflevector(lo, hi, 0, 1, 2, 3, 4, 5, 6, 7,
                                 8, 9, 10, 11, 12, 13, 14, 15);
}

__device__ __forceinline__ void mma_step_2x4(const bf16_t* Alds, const bf16_t* Blds,
                                             int wy, int wx, int lane,
                                             v8f (&acc)[2][4]) {
  const int lm = lane & 15;
  const int lh = lane >> 4;
  v16bf af[2], bf[4];
#pragma unroll
  for (int mt = 0; mt < 2; ++mt) {
    const bf16_t* r = Alds + (size_t)(wy * 32 + mt * 16 + lm) * LDS_STRIDE;
    af[mt] = make_frag(r, lh * 8, 16 + lh * 8);
  }
#pragma unroll
  for (int nt = 0; nt < 4; ++nt) {
    const bf16_t* r = Blds + (size_t)(wx * 64 + nt * 16 + lm) * LDS_STRIDE;
    bf[nt] = make_frag(r, lh * 8, 16 + lh * 8);
  }
#pragma unroll
  for (int mt = 0; mt < 2; ++mt)
#pragma unroll
    for (int nt = 0; nt < 4; ++nt)
      acc[mt][nt] = __builtin_amdgcn_wmma_f32_16x16x32_bf16(
          false, af[mt], false, bf[nt], (short)0, acc[mt][nt], false, false);
}

// ---------------------------------------------------------------------------
// GEMM1: attnL[b,q,s] = LeakyReLU( dot(query[b,q,:], context[b,s,:]) )
// Block tile 128(q) x 128(s); K = D = 1024 (32 steps of 32).
// grid = (8 = 2 qtiles * 4 stiles, B); block = 256 (8 waves, 4x2 wave grid).
// ---------------------------------------------------------------------------
__global__ __launch_bounds__(256) void gemm1_kernel(const float* __restrict__ query,
                                                    const float* __restrict__ context,
                                                    float* __restrict__ attnL) {
  __shared__ alignas(16) bf16_t Alds[128 * LDS_STRIDE];
  __shared__ alignas(16) bf16_t Blds[128 * LDS_STRIDE];

  const int b = blockIdx.y;
  const int qt = blockIdx.x >> 2;  // 0..1
  const int st = blockIdx.x & 3;   // 0..3
  const int q0 = qt * 128, s0 = st * 128;
  const float* qp = query + ((size_t)b * Q_N + q0) * D_N;
  const float* cp = context + ((size_t)b * S_N + s0) * D_N;

  const int t = threadIdx.x;
  const int lane = t & 31;
  const int w = t >> 5;
  const int wy = w >> 1, wx = w & 1;
  const int lrow = t >> 3;        // 0..31
  const int lcol = (t & 7) * 4;   // 0..28

  v8f acc[2][4];
#pragma unroll
  for (int mt = 0; mt < 2; ++mt)
#pragma unroll
    for (int nt = 0; nt < 4; ++nt)
#pragma unroll
      for (int i = 0; i < 8; ++i) acc[mt][nt][i] = 0.0f;

  for (int ks = 0; ks < D_N / 32; ++ks) {
    const int k0 = ks * 32;
    // Stage 128x32 fp32 -> bf16 LDS for both operands (both row-contiguous in K=d).
#pragma unroll
    for (int p = 0; p < 4; ++p) {
      const int row = lrow + p * 32;
      const float4 va = *(const float4*)(qp + (size_t)row * D_N + k0 + lcol);
      const float4 vb = *(const float4*)(cp + (size_t)row * D_N + k0 + lcol);
      v4bf a4 = {(bf16_t)va.x, (bf16_t)va.y, (bf16_t)va.z, (bf16_t)va.w};
      v4bf b4 = {(bf16_t)vb.x, (bf16_t)vb.y, (bf16_t)vb.z, (bf16_t)vb.w};
      *(v4bf*)&Alds[(size_t)row * LDS_STRIDE + lcol] = a4;
      *(v4bf*)&Blds[(size_t)row * LDS_STRIDE + lcol] = b4;
    }
    __syncthreads();
    mma_step_2x4(Alds, Blds, wy, wx, lane, acc);
    __syncthreads();
  }

  // Epilogue: LeakyReLU(0.1) then store fp32 (B,Q,S).
  const int lm = lane & 15, lh = lane >> 4;
#pragma unroll
  for (int mt = 0; mt < 2; ++mt) {
#pragma unroll
    for (int v = 0; v < 8; ++v) {
      const int q = q0 + wy * 32 + mt * 16 + v + 8 * lh;
      float* orow = attnL + ((size_t)b * Q_N + q) * S_N + s0 + wx * 64;
#pragma unroll
      for (int nt = 0; nt < 4; ++nt) {
        float val = acc[mt][nt][v];
        val = val > 0.0f ? val : SLOPE * val;
        orow[nt * 16 + lm] = val;
      }
    }
  }
}

// ---------------------------------------------------------------------------
// Column sum-of-squares over q for each (b, s): fully coalesced column reduce.
// grid = (S_N/256, B); block = 256.
// ---------------------------------------------------------------------------
__global__ __launch_bounds__(256) void colnorm_kernel(const float* __restrict__ attnL,
                                                      float* __restrict__ normsq) {
  const int b = blockIdx.y;
  const int s = blockIdx.x * 256 + threadIdx.x;
  const float* p = attnL + (size_t)b * Q_N * S_N + s;
  float acc = 0.0f;
#pragma unroll 4
  for (int q = 0; q < Q_N; ++q) {
    const float v = p[(size_t)q * S_N];
    acc = fmaf(v, v, acc);
  }
  normsq[(size_t)b * S_N + s] = acc;
}

// ---------------------------------------------------------------------------
// Softmax over s (one wave per (b,q) row): p = softmax(9 * x / (sqrt(nsq)+eps)).
// Writes fp32 probs in place (for attnT transpose) + bf16 probs (for GEMM2).
// grid = (B*Q/8); block = 256 (8 waves).
// ---------------------------------------------------------------------------
__global__ __launch_bounds__(256) void softmax_kernel(float* __restrict__ attn,
                                                      const float* __restrict__ normsq,
                                                      bf16_t* __restrict__ probs) {
  const int w = threadIdx.x >> 5;
  const int lane = threadIdx.x & 31;
  const int r = blockIdx.x * 8 + w;  // flat (b,q) row, r < B*Q
  const int b = r >> 8;              // r / Q_N
  float* row = attn + (size_t)r * S_N;
  const float* ns = normsq + (size_t)b * S_N;

  float vals[16];
  float m = -3.402823466e38f;
#pragma unroll
  for (int i = 0; i < 16; ++i) {
    const int s = i * 32 + lane;
    const float v = SMOOTH * row[s] / (sqrtf(ns[s]) + EPS_F);
    vals[i] = v;
    m = fmaxf(m, v);
  }
#pragma unroll
  for (int off = 16; off >= 1; off >>= 1) m = fmaxf(m, __shfl_xor(m, off, 32));

  float sum = 0.0f;
#pragma unroll
  for (int i = 0; i < 16; ++i) {
    vals[i] = __expf(vals[i] - m);
    sum += vals[i];
  }
#pragma unroll
  for (int off = 16; off >= 1; off >>= 1) sum += __shfl_xor(sum, off, 32);
  const float inv = 1.0f / sum;

  bf16_t* prow = probs + (size_t)r * S_N;
#pragma unroll
  for (int i = 0; i < 16; ++i) {
    const int s = i * 32 + lane;
    const float pv = vals[i] * inv;
    row[s] = pv;
    prow[s] = (bf16_t)pv;
  }
}

// ---------------------------------------------------------------------------
// LDS-tiled transpose (B,Q,S) -> (B,S,Q) for the attnT output.
// grid = (S/32, Q/32, B); block = (32, 8).
// ---------------------------------------------------------------------------
__global__ __launch_bounds__(256) void transpose_kernel(const float* __restrict__ attn,
                                                        float* __restrict__ attnT) {
  __shared__ float tile[32][33];
  const int b = blockIdx.z;
  const int s0 = blockIdx.x * 32;
  const int q0 = blockIdx.y * 32;
  const int tx = threadIdx.x;  // 0..31
  const int ty = threadIdx.y;  // 0..7
  const float* src = attn + (size_t)b * Q_N * S_N;
#pragma unroll
  for (int j = 0; j < 4; ++j)
    tile[ty + j * 8][tx] = src[(size_t)(q0 + ty + j * 8) * S_N + s0 + tx];
  __syncthreads();
  float* dst = attnT + (size_t)b * S_N * Q_N;
#pragma unroll
  for (int j = 0; j < 4; ++j)
    dst[(size_t)(s0 + ty + j * 8) * Q_N + q0 + tx] = tile[tx][ty + j * 8];
}

// ---------------------------------------------------------------------------
// GEMM2: weighted[b,q,d] = sum_s probs[b,q,s] * context[b,s,d]
// Block tile 128(q) x 128(d); K = S = 512 (16 steps of 32).
// A = bf16 probs (row-contiguous in K); B transposed into LDS on store.
// grid = (16 = 2 qtiles * 8 dtiles, B); block = 256.
// ---------------------------------------------------------------------------
__global__ __launch_bounds__(256) void gemm2_kernel(const bf16_t* __restrict__ probs,
                                                    const float* __restrict__ context,
                                                    float* __restrict__ outw) {
  __shared__ alignas(16) bf16_t Alds[128 * LDS_STRIDE];
  __shared__ alignas(16) bf16_t Blds[128 * LDS_STRIDE];

  const int b = blockIdx.y;
  const int qt = blockIdx.x >> 3;  // 0..1
  const int dt = blockIdx.x & 7;   // 0..7
  const int q0 = qt * 128, d0 = dt * 128;
  const bf16_t* ap = probs + ((size_t)b * Q_N + q0) * S_N;
  const float* cp = context + (size_t)b * S_N * D_N;

  const int t = threadIdx.x;
  const int lane = t & 31;
  const int w = t >> 5;
  const int wy = w >> 1, wx = w & 1;

  v8f acc[2][4];
#pragma unroll
  for (int mt = 0; mt < 2; ++mt)
#pragma unroll
    for (int nt = 0; nt < 4; ++nt)
#pragma unroll
      for (int i = 0; i < 8; ++i) acc[mt][nt][i] = 0.0f;

  for (int ks = 0; ks < S_N / 32; ++ks) {
    const int k0 = ks * 32;
    // A tile: 128 rows x 32 bf16, already bf16 and K-contiguous.
#pragma unroll
    for (int p = 0; p < 2; ++p) {
      const int row = (t >> 2) + p * 64;
      const int col = (t & 3) * 8;
      const v8bf x = *(const v8bf*)(ap + (size_t)row * S_N + k0 + col);
      *(v8bf*)&Alds[(size_t)row * LDS_STRIDE + col] = x;
    }
    // B tile: Blds[n=d][k=s] = ctx[k0+k][d0+n]; transpose on LDS store,
    // packing 2 adjacent k values so writes are ds_store_b32.
#pragma unroll
    for (int p = 0; p < 2; ++p) {
      const int kp = (t >> 5) + p * 8;   // 0..15 (k pair)
      const int n0 = (t & 31) * 4;       // 0..124
      const float4 r0 = *(const float4*)(cp + (size_t)(k0 + 2 * kp) * D_N + d0 + n0);
      const float4 r1 = *(const float4*)(cp + (size_t)(k0 + 2 * kp + 1) * D_N + d0 + n0);
      const float e0[4] = {r0.x, r0.y, r0.z, r0.w};
      const float e1[4] = {r1.x, r1.y, r1.z, r1.w};
#pragma unroll
      for (int i = 0; i < 4; ++i) {
        v2bf pr = {(bf16_t)e0[i], (bf16_t)e1[i]};
        *(v2bf*)&Blds[(size_t)(n0 + i) * LDS_STRIDE + 2 * kp] = pr;
      }
    }
    __syncthreads();
    mma_step_2x4(Alds, Blds, wy, wx, lane, acc);
    __syncthreads();
  }

  const int lm = lane & 15, lh = lane >> 4;
#pragma unroll
  for (int mt = 0; mt < 2; ++mt) {
#pragma unroll
    for (int v = 0; v < 8; ++v) {
      const int q = q0 + wy * 32 + mt * 16 + v + 8 * lh;
      float* orow = outw + ((size_t)b * Q_N + q) * D_N + d0 + wx * 64;
#pragma unroll
      for (int nt = 0; nt < 4; ++nt) orow[nt * 16 + lm] = acc[mt][nt][v];
    }
  }
}

// ---------------------------------------------------------------------------
extern "C" void kernel_launch(void* const* d_in, const int* in_sizes, int n_in,
                              void* d_out, int out_size, void* d_ws, size_t ws_size,
                              hipStream_t stream) {
  (void)in_sizes; (void)n_in; (void)out_size; (void)ws_size;
  const float* query = (const float*)d_in[0];    // (B,Q,D)
  const float* context = (const float*)d_in[1];  // (B,S,D)

  float* out_w = (float*)d_out;                           // (B,Q,D)
  float* out_attnT = out_w + (size_t)B_N * Q_N * D_N;     // (B,S,Q)

  // Workspace: attnL fp32 (B,Q,S) | normsq fp32 (B,S) | probs bf16 (B,Q,S)
  float* attnL = (float*)d_ws;
  float* normsq = attnL + (size_t)B_N * Q_N * S_N;
  bf16_t* probs = (bf16_t*)(normsq + (size_t)B_N * S_N);

  const dim3 blk(256);
  gemm1_kernel<<<dim3(8, B_N), blk, 0, stream>>>(query, context, attnL);
  colnorm_kernel<<<dim3(S_N / 256, B_N), blk, 0, stream>>>(attnL, normsq);
  softmax_kernel<<<dim3(B_N * Q_N / 8), blk, 0, stream>>>(attnL, normsq, probs);
  transpose_kernel<<<dim3(S_N / 32, Q_N / 32, B_N), dim3(32, 8), 0, stream>>>(attnL, out_attnT);
  gemm2_kernel<<<dim3(16, B_N), blk, 0, stream>>>(probs, context, out_w);
}
